// TemporalGCN_50130858279697
// MI455X (gfx1250) — matrix operand order repaired and checked by reading, not compiled, not run on hardware
//
#include <hip/hip_runtime.h>
#include <hip/hip_bf16.h>
#include <math.h>

// ---------------- problem constants (from reference) ----------------
#define TT    20     // timesteps
#define BB    4      // batch
#define NNODE 500    // nodes per sample
#define MM    2000   // BB*NNODE
#define HH    128    // hidden
#define HH2   64
#define OUTD  32
#define G4    512    // 4*HH (LSTM gates)
#define GXLD  24     // padded t-rows in transposed GX (covers v8f span 16..23)

typedef __attribute__((ext_vector_type(16))) _Float16 v16h;
typedef __attribute__((ext_vector_type(4)))  _Float16 v4h;
typedef __attribute__((ext_vector_type(8)))  float    v8f;

static __device__ __forceinline__ v8f wmma16(v16h a, v16h b, v8f c) {
  return __builtin_amdgcn_wmma_f32_16x16x32_f16(
      false, a, false, b, (short)0, c, false, false);
}

static __device__ __forceinline__ float sigm(float x) {
  return 1.0f / (1.0f + __expf(-x));
}

// Packed-fragment slot math (see analysis):
// A 16x32 tile: element (m,k): lane = m + (((k>>3)&1)<<4), slot = ((k>>4)<<3)+(k&7)
// B 32x16 tile: element (k,n): lane = n + (((k>>3)&1)<<4), slot = ((k>>4)<<3)+(k&7)

// ---------------- K0: mask ----------------
__global__ void mask_kernel(const unsigned char* __restrict__ em,
                            float* __restrict__ mask) {
  int i = blockIdx.x * 256 + threadIdx.x;
  int t = blockIdx.y;
  if (i < MM) {
    int b = i / NNODE, v = i % NNODE;
    mask[(size_t)t * MM + i] = em[((size_t)b * TT + t) * NNODE + v] ? 1.0f : 0.0f;
  }
}

// ---------------- K1: XW = positions @ W1 (K=2) ----------------
__global__ void xw_kernel(const float* __restrict__ pos,
                          const float* __restrict__ W1,
                          float* __restrict__ XW) {
  size_t idx = (size_t)blockIdx.x * 256 + threadIdx.x;
  if (idx >= (size_t)TT * MM * HH) return;
  int n = (int)(idx & (HH - 1));
  size_t ti = idx >> 7;
  XW[idx] = pos[ti * 2 + 0] * W1[n] + pos[ti * 2 + 1] * W1[HH + n];
}

// ---------------- K2: column-sum degrees -> dinv ----------------
__global__ void deg_kernel(const float* __restrict__ adj,
                           const float* __restrict__ mask,
                           float* __restrict__ dinv) {
  int i = blockIdx.x * 256 + threadIdx.x;
  int t = blockIdx.y;
  if (i >= MM) return;
  const float* At = adj + (size_t)t * MM * MM;
  const float* mt = mask + (size_t)t * MM;
  float mi = mt[i];
  float sum = 0.0f;
  if (mi != 0.0f)
    for (int j = 0; j < MM; ++j) sum += At[(size_t)j * MM + i] * mt[j];
  float deg = mi * sum + mi;
  dinv[(size_t)t * MM + i] = (deg > 0.0f) ? rsqrtf(fmaxf(deg, 1e-12f)) : 0.0f;
}

// ---------------- K3: Z = half(dinv[row] * Y) ----------------
__global__ void scalez_kernel(const float* __restrict__ Y,
                              const float* __restrict__ dinv,
                              _Float16* __restrict__ Z) {
  size_t idx = (size_t)blockIdx.x * 256 + threadIdx.x;
  if (idx >= (size_t)TT * MM * HH) return;
  Z[idx] = (_Float16)(dinv[idx >> 7] * Y[idx]);
}

// ---------------- K4/K6: GCN aggregation via WMMA ----------------
// out = f( dinv[i]*(A_mask^T @ Z + m[i]*Z[i]) + bias ); layer1=relu, layer2=*mask
__global__ __launch_bounds__(256) void gcn_agg_kernel(
    const float* __restrict__ adj, const _Float16* __restrict__ Z,
    const float* __restrict__ mask, const float* __restrict__ dinv,
    const float* __restrict__ bias, float* __restrict__ out, int layer) {
  __shared__ v16h Apk[8][32];  // [i-tile][lane] packed A fragments
  __shared__ v16h Bpk[8][32];  // [n-tile][lane] packed B fragments
  _Float16* ap = (_Float16*)Apk;
  _Float16* bp = (_Float16*)Bpk;
  const int t  = blockIdx.z;
  const int i0 = blockIdx.x * 128;
  const float*    At = adj  + (size_t)t * MM * MM;
  const _Float16* Zt = Z    + (size_t)t * MM * HH;
  const float*    mt = mask + (size_t)t * MM;
  const float*    dt = dinv + (size_t)t * MM;
  const int tid = threadIdx.x;
  const int wid = tid >> 5, lane = tid & 31, hi = lane >> 4, nr = lane & 15;

  // hoist per-thread column masks (i coordinates fixed across k loop)
  float4 mvc[4];
  int    i4c[4];
  bool   ivalid[4];
#pragma unroll
  for (int p = 0; p < 4; ++p) {
    int q = tid + p * 256;          // 0..1023
    int i4 = (q & 31) << 2;         // 0..124
    i4c[p] = i4;
    int ig = i0 + i4;
    ivalid[p] = (ig < MM);
    float4 z4 = {0, 0, 0, 0};
    mvc[p] = ivalid[p] ? *(const float4*)(mt + ig) : z4;
  }

  v8f zero = {0, 0, 0, 0, 0, 0, 0, 0};
  v8f acc[8];
#pragma unroll
  for (int q = 0; q < 8; ++q) acc[q] = zero;

  for (int k = 0; k < MM; k += 32) {
    // ---- stage A^T (masked adjacency), float4 over i, packed stores ----
#pragma unroll
    for (int p = 0; p < 4; ++p) {
      int q = tid + p * 256;
      int kk = q >> 5;              // 0..31
      int i4 = i4c[p];
      int j = k + kk;
      float4 av = {0, 0, 0, 0};
      if (j < MM && ivalid[p]) {
        av = *(const float4*)(At + (size_t)j * MM + i0 + i4);
        float mj = mt[j];
        av.x *= mj * mvc[p].x; av.y *= mj * mvc[p].y;
        av.z *= mj * mvc[p].z; av.w *= mj * mvc[p].w;
      }
      int wt = i4 >> 4, m = i4 & 15;
      int lane0 = m + (((kk >> 3) & 1) << 4);
      int slot  = ((kk >> 4) << 3) + (kk & 7);
      ap[(((wt << 5) + lane0 + 0) << 4) + slot] = (_Float16)av.x;
      ap[(((wt << 5) + lane0 + 1) << 4) + slot] = (_Float16)av.y;
      ap[(((wt << 5) + lane0 + 2) << 4) + slot] = (_Float16)av.z;
      ap[(((wt << 5) + lane0 + 3) << 4) + slot] = (_Float16)av.w;
    }
    // ---- stage Z, v4h over n, packed stores ----
#pragma unroll
    for (int p = 0; p < 4; ++p) {
      int q = tid + p * 256;
      int kk = q >> 5;
      int n4 = (q & 31) << 2;
      int j = k + kk;
      v4h zv = {(_Float16)0.f, (_Float16)0.f, (_Float16)0.f, (_Float16)0.f};
      if (j < MM) zv = *(const v4h*)(Zt + (size_t)j * HH + n4);
      int ntl = n4 >> 4;
      int lane0 = (((kk >> 3) & 1) << 4) + (n4 & 15);
      int slot  = ((kk >> 4) << 3) + (kk & 7);
      bp[(((ntl << 5) + lane0 + 0) << 4) + slot] = zv[0];
      bp[(((ntl << 5) + lane0 + 1) << 4) + slot] = zv[1];
      bp[(((ntl << 5) + lane0 + 2) << 4) + slot] = zv[2];
      bp[(((ntl << 5) + lane0 + 3) << 4) + slot] = zv[3];
    }
    __syncthreads();
    v16h af = Apk[wid][lane];       // single 32B LDS read
#pragma unroll
    for (int nt = 0; nt < 8; ++nt)
      acc[nt] = wmma16(af, Bpk[nt][lane], acc[nt]);
    __syncthreads();
  }

  const int irow = i0 + wid * 16;
#pragma unroll
  for (int nt = 0; nt < 8; ++nt) {
#pragma unroll
    for (int r = 0; r < 8; ++r) {
      int i = irow + r + (hi << 3);
      int n = nt * 16 + nr;
      if (i < MM) {
        float di = dt[i], mi = mt[i];
        float zs = (float)Zt[(size_t)i * HH + n];
        float val = di * (acc[nt][r] + mi * zs) + bias[n];
        val = (layer == 1) ? fmaxf(val, 0.0f) : val * mi;
        out[((size_t)t * MM + i) * HH + n] = val;
      }
    }
  }
}

// ---------------- K5: Z2 = half(dinv * (H1 @ W2)) ----------------
__global__ __launch_bounds__(256) void h1w2_kernel(
    const float* __restrict__ H1, const float* __restrict__ W2f,
    const float* __restrict__ dinv, _Float16* __restrict__ Z2) {
  __shared__ v16h Apk[8][32];
  __shared__ v16h Bpk[8][32];
  _Float16* ap = (_Float16*)Apk;
  _Float16* bp = (_Float16*)Bpk;
  const int t  = blockIdx.z;
  const int i0 = blockIdx.x * 128;
  const float* dt = dinv + (size_t)t * MM;
  const int tid = threadIdx.x;
  const int wid = tid >> 5, lane = tid & 31, hi = lane >> 4, nr = lane & 15;

  v8f zero = {0, 0, 0, 0, 0, 0, 0, 0};
  v8f acc[8];
#pragma unroll
  for (int q = 0; q < 8; ++q) acc[q] = zero;

  for (int k = 0; k < HH; k += 32) {
    // A tile: float4 over k -> single v4h packed store (lane const, slots consecutive)
#pragma unroll
    for (int p = 0; p < 4; ++p) {
      int q = tid + p * 256;        // 0..1023
      int il = q >> 3;              // 0..127
      int k4 = (q & 7) << 2;        // 0..28
      int i = i0 + il;
      v4h hv = {(_Float16)0.f, (_Float16)0.f, (_Float16)0.f, (_Float16)0.f};
      if (i < MM) {
        float4 xv = *(const float4*)(H1 + ((size_t)t * MM + i) * HH + k + k4);
        hv[0] = (_Float16)xv.x; hv[1] = (_Float16)xv.y;
        hv[2] = (_Float16)xv.z; hv[3] = (_Float16)xv.w;
      }
      int wt = il >> 4, m = il & 15;
      int lane0 = m + (((k4 >> 3) & 1) << 4);
      int slot  = ((k4 >> 4) << 3) + (k4 & 7);
      *(v4h*)&ap[(((wt << 5) + lane0) << 4) + slot] = hv;
    }
    // B tile: float4 over n
#pragma unroll
    for (int p = 0; p < 4; ++p) {
      int q = tid + p * 256;
      int kk = q >> 5;
      int n4 = (q & 31) << 2;
      float4 wv = *(const float4*)(W2f + (size_t)(k + kk) * HH + n4);
      int ntl = n4 >> 4;
      int lane0 = (((kk >> 3) & 1) << 4) + (n4 & 15);
      int slot  = ((kk >> 4) << 3) + (kk & 7);
      bp[(((ntl << 5) + lane0 + 0) << 4) + slot] = (_Float16)wv.x;
      bp[(((ntl << 5) + lane0 + 1) << 4) + slot] = (_Float16)wv.y;
      bp[(((ntl << 5) + lane0 + 2) << 4) + slot] = (_Float16)wv.z;
      bp[(((ntl << 5) + lane0 + 3) << 4) + slot] = (_Float16)wv.w;
    }
    __syncthreads();
    v16h af = Apk[wid][lane];
#pragma unroll
    for (int nt = 0; nt < 8; ++nt)
      acc[nt] = wmma16(af, Bpk[nt][lane], acc[nt]);
    __syncthreads();
  }

  const int irow = i0 + wid * 16;
#pragma unroll
  for (int nt = 0; nt < 8; ++nt)
#pragma unroll
    for (int r = 0; r < 8; ++r) {
      int i = irow + r + (hi << 3);
      int n = nt * 16 + nr;
      if (i < MM)
        Z2[((size_t)t * MM + i) * HH + n] = (_Float16)(dt[i] * acc[nt][r]);
    }
}

// ---------------- K7: GX2[s][g][t] = (xs @ W_ih^T + b_ih + b_hh) transposed ----------------
__global__ __launch_bounds__(256) void gatesx_kernel(
    const float* __restrict__ PH, const float* __restrict__ Wih,
    const float* __restrict__ bih, const float* __restrict__ bhh,
    float* __restrict__ GX2) {
  __shared__ v16h Apk[8][32];
  __shared__ v16h Bpk[8][32];
  _Float16* ap = (_Float16*)Apk;
  _Float16* bp = (_Float16*)Bpk;
  const int r0  = blockIdx.x * 128;
  const int g0b = blockIdx.y * 128;
  const int R = MM * TT;
  const int tid = threadIdx.x;
  const int wid = tid >> 5, lane = tid & 31, hi = lane >> 4, nr = lane & 15;

  v8f zero = {0, 0, 0, 0, 0, 0, 0, 0};
  v8f acc[8];
#pragma unroll
  for (int q = 0; q < 8; ++q) acc[q] = zero;

  for (int k = 0; k < HH; k += 32) {
    // A tile: rows of xs (= placeholder transposed), float4 over k
#pragma unroll
    for (int p = 0; p < 4; ++p) {
      int q = tid + p * 256;
      int rl = q >> 3;
      int k4 = (q & 7) << 2;
      int r = r0 + rl;
      v4h hv = {(_Float16)0.f, (_Float16)0.f, (_Float16)0.f, (_Float16)0.f};
      if (r < R) {
        int s = r / TT, tt2 = r % TT;
        float4 xv = *(const float4*)(PH + ((size_t)tt2 * MM + s) * HH + k + k4);
        hv[0] = (_Float16)xv.x; hv[1] = (_Float16)xv.y;
        hv[2] = (_Float16)xv.z; hv[3] = (_Float16)xv.w;
      }
      int wt = rl >> 4, m = rl & 15;
      int lane0 = m + (((k4 >> 3) & 1) << 4);
      int slot  = ((k4 >> 4) << 3) + (k4 & 7);
      *(v4h*)&ap[(((wt << 5) + lane0) << 4) + slot] = hv;
    }
    // B tile: W_ih^T, float4 over k -> single v4h packed store
#pragma unroll
    for (int p = 0; p < 4; ++p) {
      int q = tid + p * 256;
      int nn = q >> 3;
      int k4 = (q & 7) << 2;
      float4 wv = *(const float4*)(Wih + (size_t)(g0b + nn) * HH + k + k4);
      v4h hv = {(_Float16)wv.x, (_Float16)wv.y, (_Float16)wv.z, (_Float16)wv.w};
      int ntl = nn >> 4;
      int lane0 = (((k4 >> 3) & 1) << 4) + (nn & 15);
      int slot  = ((k4 >> 4) << 3) + (k4 & 7);
      *(v4h*)&bp[(((ntl << 5) + lane0) << 4) + slot] = hv;
    }
    __syncthreads();
    v16h af = Apk[wid][lane];
#pragma unroll
    for (int nt = 0; nt < 8; ++nt)
      acc[nt] = wmma16(af, Bpk[nt][lane], acc[nt]);
    __syncthreads();
  }

  const int rrow = r0 + wid * 16;
#pragma unroll
  for (int nt = 0; nt < 8; ++nt)
#pragma unroll
    for (int r = 0; r < 8; ++r) {
      int rr = rrow + r + (hi << 3);
      int g = g0b + nt * 16 + nr;
      if (rr < R) {
        int s = rr / TT, tt2 = rr % TT;
        GX2[((size_t)s * G4 + g) * GXLD + tt2] = acc[nt][r] + bih[g] + bhh[g];
      }
    }
}

// ---------------- K7b: W_hh^T packed in B-fragment order ----------------
__global__ void whht_kernel(const float* __restrict__ Whh,
                            _Float16* __restrict__ Wpk) {
  int idx = blockIdx.x * 256 + threadIdx.x;
  if (idx >= HH * G4) return;
  int k = idx >> 9, g = idx & (G4 - 1);
  int kt = k >> 5, kk = k & 31;
  int gt = g >> 4, nr = g & 15;
  int lane = (((kk >> 3) & 1) << 4) + nr;
  int slot = ((kk >> 4) << 3) + (kk & 7);
  Wpk[(((size_t)(kt * 32 + gt) << 5) + lane) * 16 + slot] =
      (_Float16)Whh[(size_t)g * HH + k];
}

// ---------------- K8: persistent LSTM scan (16 wave32s) ----------------
__global__ __launch_bounds__(512) void lstm_kernel(
    const float* __restrict__ GX2, const _Float16* __restrict__ Wpk,
    float* __restrict__ LO) {
  __shared__ v16h hpk[8][32];        // [mt*4+kt][lane] packed h A-fragments
  __shared__ float cst[20][128];
  __shared__ float gates[20][516];
  _Float16* hp = (_Float16*)hpk;
  const v16h* Wv = (const v16h*)Wpk;
  const int tid = threadIdx.x;
  const int wid = tid >> 5, lane = tid & 31, hi = lane >> 4, nr = lane & 15;
  const int g0 = wid * 32;           // 16 waves x 32 gate cols

  for (int idx = tid; idx < 8 * 32 * 16; idx += 512) hp[idx] = (_Float16)0.f;
  for (int idx = tid; idx < 20 * 128; idx += 512) cst[idx >> 7][idx & 127] = 0.f;
  __syncthreads();

  for (int s = 0; s < MM; ++s) {
    v8f acc[2][2];
    v8f zero = {0, 0, 0, 0, 0, 0, 0, 0};
    // C init from transposed GX: one aligned 32B vector load per tile
#pragma unroll
    for (int mt2 = 0; mt2 < 2; ++mt2)
#pragma unroll
      for (int nt = 0; nt < 2; ++nt) {
        int base = mt2 * 16 + (hi << 3);   // 0,8,16,24
        int g = g0 + nt * 16 + nr;
        acc[mt2][nt] = (base < TT)
            ? *(const v8f*)(GX2 + ((size_t)s * G4 + g) * GXLD + base)
            : zero;                         // rows 20..23 garbage are discarded below
      }
#pragma unroll
    for (int kt = 0; kt < 4; ++kt) {
      v16h af0 = hpk[kt][lane];            // mt=0
      v16h af1 = hpk[4 + kt][lane];        // mt=1
#pragma unroll
      for (int nt = 0; nt < 2; ++nt) {
        v16h bf = Wv[(size_t)(kt * 32 + (wid * 2 + nt)) * 32 + lane];
        acc[0][nt] = wmma16(af0, bf, acc[0][nt]);
        acc[1][nt] = wmma16(af1, bf, acc[1][nt]);
      }
    }
#pragma unroll
    for (int mt2 = 0; mt2 < 2; ++mt2)
#pragma unroll
      for (int nt = 0; nt < 2; ++nt)
#pragma unroll
        for (int r = 0; r < 8; ++r) {
          int trow = mt2 * 16 + r + (hi << 3);
          if (trow < TT) gates[trow][g0 + nt * 16 + nr] = acc[mt2][nt][r];
        }
    __syncthreads();
    for (int idx = tid; idx < TT * HH; idx += 512) {
      int tt2 = idx >> 7, n = idx & 127;
      float gi = gates[tt2][n];
      float gf = gates[tt2][HH + n];
      float gg = gates[tt2][2 * HH + n];
      float go = gates[tt2][3 * HH + n];
      float c = sigm(gf) * cst[tt2][n] + sigm(gi) * tanhf(gg);
      float h = sigm(go) * tanhf(c);
      cst[tt2][n] = c;
      // write h into packed A-fragment position
      int mt2 = tt2 >> 4, m = tt2 & 15;
      int kt = n >> 5, kl = n & 31;
      int lpk = m + (((kl >> 3) & 1) << 4);
      int slot = ((kl >> 4) << 3) + (kl & 7);
      hp[((((mt2 << 2) + kt) << 5) + lpk) * 16 + slot] = (_Float16)h;
      LO[((size_t)s * TT + tt2) * HH + n] = h;
    }
    __syncthreads();
  }
}

// ---------------- K9: MLP head ----------------
__global__ void head_kernel(const float* __restrict__ LO,
                            const float* __restrict__ fc1w,
                            const float* __restrict__ fc1b,
                            const float* __restrict__ fc2w,
                            const float* __restrict__ fc2b,
                            float* __restrict__ out) {
  int row = blockIdx.x * 256 + threadIdx.x;
  if (row >= MM * TT) return;
  const float* x = LO + (size_t)row * HH;
  float h2[HH2];
#pragma unroll 4
  for (int o = 0; o < HH2; ++o) {
    float a = fc1b[o];
    for (int k = 0; k < HH; ++k) a += x[k] * fc1w[(size_t)k * HH2 + o];
    h2[o] = fmaxf(a, 0.0f);
  }
#pragma unroll 4
  for (int u = 0; u < OUTD; ++u) {
    float a = fc2b[u];
    for (int o = 0; o < HH2; ++o) a += h2[o] * fc2w[(size_t)o * OUTD + u];
    out[(size_t)row * OUTD + u] = a;
  }
}

// ---------------- launch ----------------
extern "C" void kernel_launch(void* const* d_in, const int* in_sizes, int n_in,
                              void* d_out, int out_size, void* d_ws, size_t ws_size,
                              hipStream_t stream) {
  const float* positions = (const float*)d_in[0];
  const float* adjacency = (const float*)d_in[1];
  const unsigned char* ego = (const unsigned char*)d_in[2];  // bool array
  const float* W1   = (const float*)d_in[3];
  const float* b1   = (const float*)d_in[4];
  const float* W2   = (const float*)d_in[5];
  const float* b2   = (const float*)d_in[6];
  const float* W_ih = (const float*)d_in[7];
  const float* W_hh = (const float*)d_in[8];
  const float* b_ih = (const float*)d_in[9];
  const float* b_hh = (const float*)d_in[10];
  const float* fc1w = (const float*)d_in[11];
  const float* fc1b = (const float*)d_in[12];
  const float* fc2w = (const float*)d_in[13];
  const float* fc2b = (const float*)d_in[14];

  char* w = (char*)d_ws;
  auto alloc = [&](size_t bytes) -> void* {
    void* p = (void*)w;
    w += (bytes + 255) & ~(size_t)255;
    return p;
  };
  float*    mask  = (float*)alloc((size_t)TT * MM * 4);
  float*    dinv  = (float*)alloc((size_t)TT * MM * 4);
  float*    bufA  = (float*)alloc((size_t)TT * MM * HH * 4);    // XW -> H1 -> placeholder
  _Float16* bufZ  = (_Float16*)alloc((size_t)TT * MM * HH * 2); // Z1 -> Z2
  float*    GX2   = (float*)alloc((size_t)MM * G4 * GXLD * 4);  // transposed gates-x
  _Float16* Wpk   = (_Float16*)alloc((size_t)HH * G4 * 2);      // packed W_hh^T
  float*    lstmo = (float*)alloc((size_t)MM * TT * HH * 4);
  (void)ws_size; (void)n_in; (void)in_sizes; (void)out_size;

  mask_kernel<<<dim3((MM + 255) / 256, TT), 256, 0, stream>>>(ego, mask);
  xw_kernel<<<(int)(((size_t)TT * MM * HH + 255) / 256), 256, 0, stream>>>(
      positions, W1, bufA);
  deg_kernel<<<dim3((MM + 255) / 256, TT), 256, 0, stream>>>(adjacency, mask, dinv);
  scalez_kernel<<<(int)(((size_t)TT * MM * HH + 255) / 256), 256, 0, stream>>>(
      bufA, dinv, bufZ);
  gcn_agg_kernel<<<dim3((MM + 127) / 128, 1, TT), 256, 0, stream>>>(
      adjacency, bufZ, mask, dinv, b1, bufA, 1);
  h1w2_kernel<<<dim3((MM + 127) / 128, 1, TT), 256, 0, stream>>>(
      bufA, W2, dinv, bufZ);
  gcn_agg_kernel<<<dim3((MM + 127) / 128, 1, TT), 256, 0, stream>>>(
      adjacency, bufZ, mask, dinv, b2, bufA, 2);
  gatesx_kernel<<<dim3((MM * TT + 127) / 128, G4 / 128), 256, 0, stream>>>(
      bufA, W_ih, b_ih, b_hh, GX2);
  whht_kernel<<<(HH * G4 + 255) / 256, 256, 0, stream>>>(W_hh, Wpk);
  lstm_kernel<<<1, 512, 0, stream>>>(GX2, Wpk, lstmo);
  head_kernel<<<(MM * TT + 255) / 256, 256, 0, stream>>>(
      lstmo, fc1w, fc1b, fc2w, fc2b, (float*)d_out);
}
